// TransformerEncoder_9826885173944
// MI455X (gfx1250) — matrix operand order repaired and checked by reading, not compiled, not run on hardware
//
#include <hip/hip_runtime.h>
#include <hip/hip_bf16.h>

typedef _Float16 f16;
typedef __attribute__((ext_vector_type(16))) _Float16 v16h;
typedef __attribute__((ext_vector_type(8)))  _Float16 v8h;
typedef __attribute__((ext_vector_type(2)))  _Float16 v2h;
typedef __attribute__((ext_vector_type(8)))  float    v8f;
typedef __attribute__((ext_vector_type(2)))  float    v2f;
typedef int v4i __attribute__((vector_size(16)));   // matches builtin proto V4i

#define AS1 __attribute__((address_space(1)))
#define AS3 __attribute__((address_space(3)))

#if defined(__gfx1250__) && __has_builtin(__builtin_amdgcn_global_load_async_to_lds_b128) && __has_builtin(__builtin_amdgcn_s_wait_asynccnt)
#define HAS_ASYNC 1
#else
#define HAS_ASYNC 0
#endif

// ---------------- problem constants ----------------
constexpr int DIMC   = 768;
constexpr int DEPTH  = 6;
constexpr int HEADS  = 12;
constexpr int DHEAD  = 64;
constexpr int INNER  = HEADS * DHEAD;   // 768
constexpr int MLP    = 4 * DIMC;        // 3072
constexpr int BB     = 4;
constexpr int NN     = 2048;
constexpr int MTOK   = BB * NN;         // 8192 tokens
constexpr float EPSL = 1e-5f;

// ---------------- helpers ----------------
static __device__ inline v8f wmma_f16(v16h a, v16h b, v8f c) {
  return __builtin_amdgcn_wmma_f32_16x16x32_f16(false, a, false, b, (short)0, c,
                                                false, false);
}
static __device__ inline v16h cat16(v8h lo, v8h hi) {
  return __builtin_shufflevector(lo, hi, 0, 1, 2, 3, 4, 5, 6, 7, 8, 9, 10, 11,
                                 12, 13, 14, 15);
}
static __device__ inline float gelu_exact(float v) {
  return 0.5f * v * (1.0f + erff(v * 0.70710678118654752f));
}
// 16-byte global->LDS copy: async (ASYNCcnt-tracked) path when available
static __device__ inline void copy16(void* lds, const void* g) {
#if HAS_ASYNC
  __builtin_amdgcn_global_load_async_to_lds_b128((AS1 v4i*)g, (AS3 v4i*)lds,
                                                 0, 0);
#else
  *(v8h*)lds = *(const v8h*)g;
#endif
}
static __device__ inline void wait_async() {
#if HAS_ASYNC
  __builtin_amdgcn_s_wait_asynccnt(0);
#endif
}

// ---------------- LayerNorm: fp32 -> f16 (or fp32 for final) ----------------
template <bool OUT16>
__global__ void ln_kernel(const float* __restrict__ X,
                          const float* __restrict__ g,
                          const float* __restrict__ b,
                          f16* __restrict__ o16, float* __restrict__ o32) {
  const int row = blockIdx.x;
  const int tid = threadIdx.x;
  const float* xr = X + (size_t)row * DIMC;

  float x0 = xr[tid], x1 = xr[tid + 256], x2 = xr[tid + 512];
  float s = x0 + x1 + x2;
  float q = x0 * x0 + x1 * x1 + x2 * x2;
#pragma unroll
  for (int m = 1; m < 32; m <<= 1) {
    s += __shfl_xor(s, m, 32);
    q += __shfl_xor(q, m, 32);
  }
  __shared__ float red[2][8];
  const int wid = tid >> 5, lane = tid & 31;
  if (lane == 0) { red[0][wid] = s; red[1][wid] = q; }
  __syncthreads();
  s = 0.f; q = 0.f;
#pragma unroll
  for (int j = 0; j < 8; ++j) { s += red[0][j]; q += red[1][j]; }
  const float mu = s * (1.0f / DIMC);
  const float var = q * (1.0f / DIMC) - mu * mu;
  const float rstd = rsqrtf(var + EPSL);

#pragma unroll
  for (int j = 0; j < 3; ++j) {
    int c = tid + j * 256;
    float xv = (j == 0) ? x0 : (j == 1) ? x1 : x2;
    float y = (xv - mu) * rstd * g[c] + b[c];
    if (OUT16) o16[(size_t)row * DIMC + c] = (f16)y;
    else       o32[(size_t)row * DIMC + c] = y;
  }
}

// ---------------- WMMA GEMM: A[M,K] f16  x  B[K,N] f32(->f16) ----------------
// EPI: 0 = store f16 (no bias)        1 = bias + exact GELU, store f16
//      2 = fp32 residual += acc       3 = fp32 residual += acc + bias
// double-buffered BK=32 pipeline, one barrier per k-step
template <int EPI>
__global__ void __launch_bounds__(256)
gemm_wmma(const f16* __restrict__ A, const float* __restrict__ Bw,
          const float* __restrict__ bias, f16* __restrict__ outH,
          float* __restrict__ outF, int M, int Nc, int K) {
  __shared__ f16 As[2][128][32];   // [m][k]
  __shared__ f16 Bs[2][128][32];   // [n][k]  (transposed for B-fragment loads)

  const int tid = threadIdx.x;
  const int wid = tid >> 5, lane = tid & 31;
  const int lm = lane & 15, lg = lane >> 4;
  const int wave_m = wid & 3;       // 4 waves over M (32 rows each)
  const int wave_n = wid >> 2;      // 2 waves over N (64 cols each)
  const int m0 = blockIdx.y * 128, n0 = blockIdx.x * 128;

  // B staging map: each thread owns 2 n-cols x 8 k-rows
  const int nsub = (tid & 63) * 2;  // 0..126
  const int ksub = (tid >> 6) * 8;  // 0,8,16,24

  v8f acc[2][4];
#pragma unroll
  for (int i = 0; i < 2; ++i)
#pragma unroll
    for (int j = 0; j < 4; ++j) acc[i][j] = (v8f)(0.0f);

  auto stage = [&](int buf, int k0) {
    // A tile: 128x32 f16, 16B chunks (async path)
#pragma unroll
    for (int c = tid; c < 512; c += 256) {
      int r = c >> 2, kc = (c & 3) * 8;
      copy16(&As[buf][r][kc], &A[(size_t)(m0 + r) * K + k0 + kc]);
    }
    // B tile: fp32 -> f16, transpose via register pack, 16B LDS stores
    const float* bp = &Bw[(size_t)(k0 + ksub) * Nc + n0 + nsub];
    v8h c0, c1;
#pragma unroll
    for (int kk = 0; kk < 8; ++kk) {
      v2f t = *(const v2f*)(bp + (size_t)kk * Nc);
      c0[kk] = (f16)t.x;
      c1[kk] = (f16)t.y;
    }
    *(v8h*)&Bs[buf][nsub][ksub]     = c0;
    *(v8h*)&Bs[buf][nsub + 1][ksub] = c1;
    if (k0 + 96 <= K)  // prefetch weight tile two steps ahead
      __builtin_prefetch(bp + (size_t)64 * Nc, 0, 1);
  };

  stage(0, 0);
  int cur = 0;
  const int nk = K / 32;
  for (int kt = 0; kt < nk; ++kt) {
    wait_async();
    __syncthreads();
    if (kt + 1 < nk) stage(1 - cur, (kt + 1) * 32);

    v16h af[2], bf[4];
#pragma unroll
    for (int mt = 0; mt < 2; ++mt) {
      int r = wave_m * 32 + mt * 16 + lm;
      v8h lo = *(const v8h*)&As[cur][r][lg * 8];
      v8h hi = *(const v8h*)&As[cur][r][lg * 8 + 16];
      af[mt] = cat16(lo, hi);
    }
#pragma unroll
    for (int nt = 0; nt < 4; ++nt) {
      int c = wave_n * 64 + nt * 16 + lm;
      v8h lo = *(const v8h*)&Bs[cur][c][lg * 8];
      v8h hi = *(const v8h*)&Bs[cur][c][lg * 8 + 16];
      bf[nt] = cat16(lo, hi);
    }
#pragma unroll
    for (int mt = 0; mt < 2; ++mt)
#pragma unroll
      for (int nt = 0; nt < 4; ++nt)
        acc[mt][nt] = wmma_f16(af[mt], bf[nt], acc[mt][nt]);
    cur ^= 1;
  }

  // epilogue  (row m = i + 8*lg within a 16-row subtile, col n = lm)
#pragma unroll
  for (int mt = 0; mt < 2; ++mt)
#pragma unroll
    for (int nt = 0; nt < 4; ++nt) {
      const int n = n0 + wave_n * 64 + nt * 16 + lm;
      float bv = (EPI == 1 || EPI == 3) ? bias[n] : 0.0f;
#pragma unroll
      for (int i = 0; i < 8; ++i) {
        const int m = m0 + wave_m * 32 + mt * 16 + lg * 8 + i;
        float v = acc[mt][nt][i] + bv;
        if (EPI == 0)      outH[(size_t)m * Nc + n] = (f16)v;
        else if (EPI == 1) outH[(size_t)m * Nc + n] = (f16)gelu_exact(v);
        else               outF[(size_t)m * Nc + n] += v;
      }
    }
}

// ---------------- Flash attention ----------------
// grid: (N/128, B*HEADS); 8 waves x 16 query rows per block
__global__ void __launch_bounds__(256)
attn_kernel(const f16* __restrict__ QKV, f16* __restrict__ ATT) {
  __shared__ f16 Ks[64][64];      // [key][d]
  __shared__ f16 Vs[64][64];      // [d][key]  (transposed)
  __shared__ f16 Ps[8][16][64];   // per-wave probability tile

  const int tid = threadIdx.x;
  const int wid = tid >> 5, lane = tid & 31;
  const int lm = lane & 15, lg = lane >> 4;
  const int q0 = blockIdx.x * 128;
  const int b = blockIdx.y / HEADS, h = blockIdx.y % HEADS;
  const float scale = 0.125f;  // 1/sqrt(64)

  const f16* qkv_b = QKV + (size_t)b * NN * (3 * INNER);

  // persistent Q fragments: row q = q0 + wid*16 + lm
  v16h qf[2];
  {
    const int q = q0 + wid * 16 + lm;
    const f16* qp = qkv_b + (size_t)q * (3 * INNER) + h * DHEAD;
#pragma unroll
    for (int kd = 0; kd < 2; ++kd) {
      v8h lo = *(const v8h*)(qp + kd * 32 + lg * 8);
      v8h hi = *(const v8h*)(qp + kd * 32 + lg * 8 + 16);
      qf[kd] = cat16(lo, hi);
    }
  }

  v8f o[4];
#pragma unroll
  for (int nt = 0; nt < 4; ++nt) o[nt] = (v8f)(0.0f);
  float mrow[8], lrow[8];
#pragma unroll
  for (int i = 0; i < 8; ++i) { mrow[i] = -1e30f; lrow[i] = 0.0f; }

  const int key  = tid >> 2;        // K staging: 0..63
  const int dseg = (tid & 3) * 16;  // 0,16,32,48
  const int vkp  = (tid & 31) * 2;  // V staging: key pair
  const int vds  = (tid >> 5) * 8;  // 8 d-values

  for (int kb = 0; kb < NN / 64; ++kb) {
    const int kstart = kb * 64;
    __syncthreads();
    // K tile [key][d]: straight 16B copies (async path)
    {
      const f16* kr = qkv_b + (size_t)(kstart + key) * (3 * INNER) + INNER +
                      h * DHEAD + dseg;
      copy16(&Ks[key][dseg], kr);
      copy16(&Ks[key][dseg + 8], kr + 8);
    }
    // V tile transposed [d][key]: pack key-pairs, 4B LDS stores
    {
      const f16* vr = qkv_b + (size_t)(kstart + vkp) * (3 * INNER) +
                      2 * INNER + h * DHEAD + vds;
      v8h a = *(const v8h*)vr;
      v8h c = *(const v8h*)(vr + 3 * INNER);
#pragma unroll
      for (int j = 0; j < 8; ++j)
        *(v2h*)&Vs[vds + j][vkp] = (v2h){a[j], c[j]};
    }
    wait_async();
    __syncthreads();

    // S = Q K^T  (contraction over d = 64, two k-steps)
    v8f s[4];
#pragma unroll
    for (int nt = 0; nt < 4; ++nt) {
      s[nt] = (v8f)(0.0f);
#pragma unroll
      for (int kd = 0; kd < 2; ++kd) {
        v8h lo = *(const v8h*)&Ks[nt * 16 + lm][kd * 32 + lg * 8];
        v8h hi = *(const v8h*)&Ks[nt * 16 + lm][kd * 32 + lg * 8 + 16];
        s[nt] = wmma_f16(qf[kd], cat16(lo, hi), s[nt]);
      }
    }
#pragma unroll
    for (int nt = 0; nt < 4; ++nt)
#pragma unroll
      for (int i = 0; i < 8; ++i) s[nt][i] *= scale;

    // online softmax (per-row stats replicated across 16-lane group)
#pragma unroll
    for (int i = 0; i < 8; ++i) {
      float v = s[0][i];
      v = fmaxf(v, s[1][i]); v = fmaxf(v, s[2][i]); v = fmaxf(v, s[3][i]);
      v = fmaxf(v, __shfl_xor(v, 1, 32));
      v = fmaxf(v, __shfl_xor(v, 2, 32));
      v = fmaxf(v, __shfl_xor(v, 4, 32));
      v = fmaxf(v, __shfl_xor(v, 8, 32));
      const float mnew = fmaxf(mrow[i], v);
      const float corr = __expf(mrow[i] - mnew);
      float psum = 0.f;
#pragma unroll
      for (int nt = 0; nt < 4; ++nt) {
        float p = __expf(s[nt][i] - mnew);
        s[nt][i] = p;
        psum += p;
      }
      psum += __shfl_xor(psum, 1, 32);
      psum += __shfl_xor(psum, 2, 32);
      psum += __shfl_xor(psum, 4, 32);
      psum += __shfl_xor(psum, 8, 32);
      lrow[i] = lrow[i] * corr + psum;
      mrow[i] = mnew;
#pragma unroll
      for (int nt = 0; nt < 4; ++nt) o[nt][i] *= corr;
    }

    // P -> LDS (C-layout) then reload as A-fragments
#pragma unroll
    for (int nt = 0; nt < 4; ++nt)
#pragma unroll
      for (int i = 0; i < 8; ++i)
        Ps[wid][lg * 8 + i][nt * 16 + lm] = (f16)s[nt][i];
    __syncthreads();

    v16h pf[2];
#pragma unroll
    for (int kd = 0; kd < 2; ++kd) {
      v8h lo = *(const v8h*)&Ps[wid][lm][kd * 32 + lg * 8];
      v8h hi = *(const v8h*)&Ps[wid][lm][kd * 32 + lg * 8 + 16];
      pf[kd] = cat16(lo, hi);
    }
#pragma unroll
    for (int nt = 0; nt < 4; ++nt)
#pragma unroll
      for (int kd = 0; kd < 2; ++kd) {
        v8h lo = *(const v8h*)&Vs[nt * 16 + lm][kd * 32 + lg * 8];
        v8h hi = *(const v8h*)&Vs[nt * 16 + lm][kd * 32 + lg * 8 + 16];
        o[nt] = wmma_f16(pf[kd], cat16(lo, hi), o[nt]);
      }
  }

  // write out  ATT[b, q, h*64 + d]
#pragma unroll
  for (int nt = 0; nt < 4; ++nt)
#pragma unroll
    for (int i = 0; i < 8; ++i) {
      const int q = q0 + wid * 16 + lg * 8 + i;
      const int d = nt * 16 + lm;
      ATT[((size_t)b * NN + q) * INNER + h * DHEAD + d] =
          (f16)(o[nt][i] / lrow[i]);
    }
}

// ---------------- host side ----------------
extern "C" void kernel_launch(void* const* d_in, const int* in_sizes, int n_in,
                              void* d_out, int out_size, void* d_ws,
                              size_t ws_size, hipStream_t stream) {
  (void)in_sizes; (void)n_in; (void)out_size; (void)ws_size;

  const float* x_in  = (const float*)d_in[0];
  const float* ln1_g = (const float*)d_in[1];
  const float* ln1_b = (const float*)d_in[2];
  const float* wqkv  = (const float*)d_in[3];
  const float* wout  = (const float*)d_in[4];
  const float* ln2_g = (const float*)d_in[5];
  const float* ln2_b = (const float*)d_in[6];
  const float* w1    = (const float*)d_in[7];
  const float* b1    = (const float*)d_in[8];
  const float* w2    = (const float*)d_in[9];
  const float* b2    = (const float*)d_in[10];
  const float* fn_g  = (const float*)d_in[11];
  const float* fn_b  = (const float*)d_in[12];

  uint8_t* ws = (uint8_t*)d_ws;
  size_t off = 0;
  auto take = [&](size_t bytes) -> void* {
    void* p = ws + off;
    off += (bytes + 255) & ~(size_t)255;
    return p;
  };
  float* X    = (float*)take((size_t)MTOK * DIMC * sizeof(float));
  f16*   LN16 = (f16*)  take((size_t)MTOK * DIMC * sizeof(f16));
  f16*   QKVb = (f16*)  take((size_t)MTOK * 3 * INNER * sizeof(f16));
  f16*   ATTb = (f16*)  take((size_t)MTOK * INNER * sizeof(f16));
  f16*   H16  = (f16*)  take((size_t)MTOK * MLP * sizeof(f16));

  (void)hipMemcpyAsync(X, x_in, (size_t)MTOK * DIMC * sizeof(float),
                       hipMemcpyDeviceToDevice, stream);

  const dim3 blk(256);
  for (int i = 0; i < DEPTH; ++i) {
    // ---- attention block ----
    ln_kernel<true><<<MTOK, blk, 0, stream>>>(X, ln1_g + i * DIMC,
                                              ln1_b + i * DIMC, LN16, nullptr);
    gemm_wmma<0><<<dim3(3 * INNER / 128, MTOK / 128), blk, 0, stream>>>(
        LN16, wqkv + (size_t)i * DIMC * 3 * INNER, nullptr, QKVb, nullptr,
        MTOK, 3 * INNER, DIMC);
    attn_kernel<<<dim3(NN / 128, BB * HEADS), blk, 0, stream>>>(QKVb, ATTb);
    gemm_wmma<2><<<dim3(DIMC / 128, MTOK / 128), blk, 0, stream>>>(
        ATTb, wout + (size_t)i * INNER * DIMC, nullptr, nullptr, X, MTOK, DIMC,
        INNER);
    // ---- MLP block ----
    ln_kernel<true><<<MTOK, blk, 0, stream>>>(X, ln2_g + i * DIMC,
                                              ln2_b + i * DIMC, LN16, nullptr);
    gemm_wmma<1><<<dim3(MLP / 128, MTOK / 128), blk, 0, stream>>>(
        LN16, w1 + (size_t)i * DIMC * MLP, b1 + (size_t)i * MLP, H16, nullptr,
        MTOK, MLP, DIMC);
    gemm_wmma<3><<<dim3(DIMC / 128, MTOK / 128), blk, 0, stream>>>(
        H16, w2 + (size_t)i * MLP * DIMC, b2 + (size_t)i * DIMC, nullptr, X,
        MTOK, DIMC, MLP);
  }
  ln_kernel<false><<<MTOK, blk, 0, stream>>>(X, fn_g, fn_b, nullptr,
                                             (float*)d_out);
}
